// AttentionBlock_27358941675958
// MI455X (gfx1250) — matrix operand order repaired and checked by reading, not compiled, not run on hardware
//
#include <hip/hip_runtime.h>
#include <hip/hip_bf16.h>

typedef __attribute__((ext_vector_type(16))) __bf16 v16bf;
typedef __attribute__((ext_vector_type(8)))  __bf16 v8bf;
typedef __attribute__((ext_vector_type(8)))  float  v8f;
typedef __attribute__((ext_vector_type(4)))  float  v4f;
typedef __attribute__((ext_vector_type(4)))  unsigned int u32x4;
typedef __attribute__((ext_vector_type(8)))  int i32x8;
typedef __attribute__((ext_vector_type(4)))  int i32x4;

#define NB 8
#define NT 4096
#define NC 768
#define ND 64
#define SLICE (3 * 64 * 32)   // one kc-slice of all three W^T matrices (bf16 elements)

__device__ __forceinline__ v8f zero8() {
  v8f z;
#pragma unroll
  for (int i = 0; i < 8; ++i) z[i] = 0.0f;
  return z;
}

// B-fragment (bf16 32x16): lane n (0-15) -> column n, K = {0..7, 16..23} ; lane n+16 -> column n, K = {8..15, 24..31}.
// p must point at columnRowBase + kbase + (lane>>4)*8 ; chunks at +0 and +16 (two b128 loads).
__device__ __forceinline__ v16bf load_bfrag(const __bf16* p) {
  v8bf lo = *(const v8bf*)(p);
  v8bf hi = *(const v8bf*)(p + 16);
  v16bf r;
#pragma unroll
  for (int e = 0; e < 8; ++e) { r[e] = lo[e]; r[e + 8] = hi[e]; }
  return r;
}

// A-fragment (bf16 16x32): lane m (0-15) -> row m, K = {0..7, 16..23}; lane m+16 -> row m, K = {8..15, 24..31}.
// p must point at rowBase + kbase + (lane>>4)*8 ; chunks at +0 and +16.
__device__ __forceinline__ v16bf load_afrag(const __bf16* p) {
  v8bf c0 = *(const v8bf*)(p);
  v8bf c1 = *(const v8bf*)(p + 16);
  v16bf r;
#pragma unroll
  for (int e = 0; e < 8; ++e) { r[e] = c0[e]; r[e + 8] = c1[e]; }
  return r;
}

// Same A-fragment pattern but sourced from fp32 memory, converted to bf16 in-register.
__device__ __forceinline__ v16bf load_afrag_f32(const float* p) {
  v4f c0 = *(const v4f*)(p);
  v4f c1 = *(const v4f*)(p + 4);
  v4f c2 = *(const v4f*)(p + 16);
  v4f c3 = *(const v4f*)(p + 20);
  v16bf r;
#pragma unroll
  for (int e = 0; e < 4; ++e) {
    r[e]      = (__bf16)c0[e];
    r[e + 4]  = (__bf16)c1[e];
    r[e + 8]  = (__bf16)c2[e];
    r[e + 12] = (__bf16)c3[e];
  }
  return r;
}

// TDM: DMA a [192 rows x 32 cols] bf16 tile of the combined W^T tensor (192 x 768,
// row stride 768) from global into LDS at lds_off. 2D descriptor, data_size = 2B.
__device__ __forceinline__ void tdm_load_slice(unsigned lds_off, unsigned long long gaddr) {
  u32x4 g0;
  g0[0] = 1u;                                             // count=1 (valid), user mode
  g0[1] = lds_off;                                        // lds_addr (bytes)
  g0[2] = (unsigned)(gaddr & 0xffffffffu);                // global_addr[31:0]
  g0[3] = (unsigned)((gaddr >> 32) & 0x01ffffffu)         // global_addr[56:32]
        | (2u << 30);                                     // type = 2 ("image")
  i32x8 g1;
  g1[0] = 1 << 16;                                        // wg_mask=0, data_size=1 (2 bytes)
  g1[1] = (int)((768u & 0xffffu) << 16);                  // tensor_dim0[15:0] -> bits 63:48
  g1[2] = (int)((768u >> 16) | ((192u & 0xffffu) << 16)); // tensor_dim0[31:16], tensor_dim1[15:0]
  g1[3] = (int)((192u >> 16) | (32u << 16));              // tensor_dim1[31:16], tile_dim0=32
  g1[4] = (int)(192u);                                    // tile_dim1=192, tile_dim2=0
  g1[5] = (int)(768u);                                    // tensor_dim0_stride[31:0]
  g1[6] = 0;                                              // stride0[47:32], stride1[15:0]
  g1[7] = 0;
  i32x4 gz;
  gz[0] = 0; gz[1] = 0; gz[2] = 0; gz[3] = 0;             // groups 2/3 unused (2D tile)
#if __clang_major__ >= 23
  i32x8 gz8;
#pragma unroll
  for (int i = 0; i < 8; ++i) gz8[i] = 0;
  __builtin_amdgcn_tensor_load_to_lds(g0, g1, gz, gz, gz8, 0);
#else
  __builtin_amdgcn_tensor_load_to_lds(g0, g1, gz, gz, 0);
#endif
}

// W [C x D] fp32 -> Wt [D x C] bf16 (so GEMM B-fragments are contiguous row loads)
__global__ void wT_kernel(const float* __restrict__ w, __bf16* __restrict__ wt) {
  int i = blockIdx.x * blockDim.x + threadIdx.x;
  if (i >= NC * ND) return;
  int c = i / ND, d = i % ND;
  wt[(size_t)d * NC + c] = (__bf16)w[i];
}

// QKV projection: 4 waves per block, each wave owns one 16-row tile of x.
// The per-kc slice of all three W^T matrices is TDM-staged into LDS (double
// buffered, prefetched one iteration ahead by wave 0) and shared by all waves.
// q is pre-scaled by D^-0.5; v is stored transposed [B][D][T].
__global__ void __launch_bounds__(128)
qkv_kernel(const float* __restrict__ x,
           const __bf16* __restrict__ wqt,   // base of contiguous [3][64][768] W^T
           const float* __restrict__ bq, const float* __restrict__ bk,
           const float* __restrict__ bv,
           __bf16* __restrict__ qo, __bf16* __restrict__ ko,
           __bf16* __restrict__ vto) {
  __shared__ __bf16 wslc[2 * SLICE];   // 24 KB: double-buffered weight slice

  const int lane = threadIdx.x & 31;
  const int wave = threadIdx.x >> 5;
  const int tile = blockIdx.x * 4 + wave;   // 16-row tile, 0..2047
  const int row0 = tile * 16;               // global row in [0, B*T)
  const int m    = lane & 15;
  const int hi   = lane >> 4;

  const unsigned lds0 = (unsigned)(size_t)(&wslc[0]);
  const unsigned lds1 = (unsigned)(size_t)(&wslc[SLICE]);
  const unsigned long long wbase = (unsigned long long)(size_t)wqt;

  v8f aq[4], ak[4], av[4];
#pragma unroll
  for (int n = 0; n < 4; ++n) { aq[n] = zero8(); ak[n] = zero8(); av[n] = zero8(); }

  if (wave == 0) tdm_load_slice(lds0, wbase);   // prologue: slice for kc = 0

  for (int kc = 0, it = 0; kc < NC; kc += 32, ++it) {
    __syncthreads();   // everyone done reading the buffer we are about to overwrite
    if (wave == 0) {
      if (kc + 32 < NC) {
        tdm_load_slice((it & 1) ? lds0 : lds1,
                       wbase + (unsigned long long)(kc + 32) * 2u);
        __builtin_amdgcn_s_wait_tensorcnt(1);   // current slice complete; prefetch in flight
      } else {
        __builtin_amdgcn_s_wait_tensorcnt(0);
      }
    }
    __syncthreads();   // publish current slice to all waves

    const int sb = (it & 1) ? SLICE : 0;        // current buffer base (elements)
    v16bf a = load_afrag_f32(x + (size_t)(row0 + m) * NC + kc + hi * 8);
#pragma unroll
    for (int n = 0; n < 4; ++n) {
      const int r = (n * 16 + m) * 32 + hi * 8;      // row offset within slice
      v16bf bq_ = load_bfrag(&wslc[sb + 0 * 64 * 32 + r]);
      aq[n] = __builtin_amdgcn_wmma_f32_16x16x32_bf16(false, a, false, bq_, (short)0, aq[n], false, false);
      v16bf bk_ = load_bfrag(&wslc[sb + 1 * 64 * 32 + r]);
      ak[n] = __builtin_amdgcn_wmma_f32_16x16x32_bf16(false, a, false, bk_, (short)0, ak[n], false, false);
      v16bf bv_ = load_bfrag(&wslc[sb + 2 * 64 * 32 + r]);
      av[n] = __builtin_amdgcn_wmma_f32_16x16x32_bf16(false, a, false, bv_, (short)0, av[n], false, false);
    }
  }

  const int b  = row0 / NT;
  const int tr = row0 % NT;
#pragma unroll
  for (int n = 0; n < 4; ++n) {
    const int col = n * 16 + m;
    const float vbq = bq[col], vbk = bk[col], vbv = bv[col];
#pragma unroll
    for (int j = 0; j < 8; ++j) {
      const int r = j + 8 * hi;   // C-layout: VGPR j -> row j (lanes 0-15) / j+8 (lanes 16-31)
      qo[(size_t)(row0 + r) * ND + col] = (__bf16)((aq[n][j] + vbq) * 0.125f);
      ko[(size_t)(row0 + r) * ND + col] = (__bf16)(ak[n][j] + vbk);
      vto[((size_t)b * ND + col) * NT + tr + r] = (__bf16)(av[n][j] + vbv);
    }
  }
}

// Flash attention, one wave per 16-query tile, 32-key blocks.
// Computes S^T = K * Q^T so that exp(S-m) lands directly in the A-fragment layout for P@V.
__global__ void __launch_bounds__(128)
attn_kernel(const __bf16* __restrict__ q, const __bf16* __restrict__ k,
            const __bf16* __restrict__ vt, float* __restrict__ y) {
  const int lane = threadIdx.x & 31;
  const int wave = threadIdx.x >> 5;
  const int tile = blockIdx.x * 4 + wave;   // 0..2047
  const int b    = tile >> 8;               // 256 tiles per batch
  const int q0   = (tile & 255) * 16;       // query base within batch
  const int m    = lane & 15;
  const int hi   = lane >> 4;

  const size_t qrow = (size_t)b * NT + q0;

  // Q as B-matrix (column = query), preloaded for both D K-steps.
  const __bf16* qp = q + (qrow + m) * ND + hi * 8;
  v16bf qb0 = load_bfrag(qp);
  v16bf qb1 = load_bfrag(qp + 32);

  v8f o[4];
#pragma unroll
  for (int n = 0; n < 4; ++n) o[n] = zero8();
  float mrun = -3.0e38f;
  float lrun = 0.0f;

  for (int kv = 0; kv <= q0; kv += 32) {
    // S^T for 32 keys: two 16-key tiles, contraction over D=64 in two WMMAs each.
    v8f st[2];
#pragma unroll
    for (int t = 0; t < 2; ++t) {
      st[t] = zero8();
      const __bf16* kp = k + ((size_t)b * NT + kv + t * 16 + m) * ND + hi * 8;
      v16bf ka0 = load_afrag(kp);
      st[t] = __builtin_amdgcn_wmma_f32_16x16x32_bf16(false, ka0, false, qb0, (short)0, st[t], false, false);
      v16bf ka1 = load_afrag(kp + 32);
      st[t] = __builtin_amdgcn_wmma_f32_16x16x32_bf16(false, ka1, false, qb1, (short)0, st[t], false, false);
    }

    // Causal mask (only near the diagonal). Row of S^T = key, column = query.
    if (kv + 31 > q0) {
      const int qabs = q0 + m;
#pragma unroll
      for (int t = 0; t < 2; ++t)
#pragma unroll
        for (int j = 0; j < 8; ++j) {
          const int kabs = kv + t * 16 + j + 8 * hi;
          if (kabs > qabs) st[t][j] = -1.0e30f;
        }
    }

    // Online softmax: each lane holds 16 of its query's 32 scores; the other 16 are in lane^16.
    float rmax = -3.0e38f;
#pragma unroll
    for (int t = 0; t < 2; ++t)
#pragma unroll
      for (int j = 0; j < 8; ++j) rmax = fmaxf(rmax, st[t][j]);
    rmax = fmaxf(rmax, __shfl_xor(rmax, 16, 32));
    const float mnew  = fmaxf(mrun, rmax);
    const float alpha = __expf(mrun - mnew);
    mrun = mnew;

    // exp + pack: S^T C-layout is exactly the A-fragment layout for P (16 queries x 32 keys).
    float psum = 0.0f;
    v16bf pa;
#pragma unroll
    for (int t = 0; t < 2; ++t)
#pragma unroll
      for (int j = 0; j < 8; ++j) {
        const float p = __expf(st[t][j] - mnew);
        psum += p;
        pa[t * 8 + j] = (__bf16)p;
      }
    psum += __shfl_xor(psum, 16, 32);
    lrun = lrun * alpha + psum;

    // Rescale O by per-row alpha (row r's stats live in lane r).
    float alp[8];
#pragma unroll
    for (int j = 0; j < 8; ++j) alp[j] = __shfl(alpha, j + 8 * hi, 32);
#pragma unroll
    for (int n = 0; n < 4; ++n)
#pragma unroll
      for (int j = 0; j < 8; ++j) o[n][j] *= alp[j];

    // O += P @ V  (V transposed in memory -> contiguous B-fragment loads)
#pragma unroll
    for (int n = 0; n < 4; ++n) {
      const __bf16* vp = vt + ((size_t)b * ND + n * 16 + m) * NT + kv + hi * 8;
      v16bf vb = load_bfrag(vp);
      o[n] = __builtin_amdgcn_wmma_f32_16x16x32_bf16(false, pa, false, vb, (short)0, o[n], false, false);
    }
  }

  // Normalize by per-row 1/l and store fp32 output.
  const float linv = 1.0f / lrun;
  float li[8];
#pragma unroll
  for (int j = 0; j < 8; ++j) li[j] = __shfl(linv, j + 8 * hi, 32);
#pragma unroll
  for (int n = 0; n < 4; ++n) {
    const int col = n * 16 + m;
#pragma unroll
    for (int j = 0; j < 8; ++j)
      y[(qrow + j + 8 * hi) * ND + col] = o[n][j] * li[j];
  }
}

extern "C" void kernel_launch(void* const* d_in, const int* in_sizes, int n_in,
                              void* d_out, int out_size, void* d_ws, size_t ws_size,
                              hipStream_t stream) {
  const float* x  = (const float*)d_in[0];
  const float* Wq = (const float*)d_in[1];
  const float* bq = (const float*)d_in[2];
  const float* Wk = (const float*)d_in[3];
  const float* bk = (const float*)d_in[4];
  const float* Wv = (const float*)d_in[5];
  const float* bv = (const float*)d_in[6];
  float* y = (float*)d_out;

  // Workspace (bf16): 3 transposed weights (contiguous -> one TDM tensor) + q + k + v^T
  __bf16* ws  = (__bf16*)d_ws;
  __bf16* wqt = ws;
  __bf16* wkt = wqt + (size_t)NC * ND;
  __bf16* wvt = wkt + (size_t)NC * ND;
  __bf16* qb  = wvt + (size_t)NC * ND;
  __bf16* kb  = qb + (size_t)NB * NT * ND;
  __bf16* vtb = kb + (size_t)NB * NT * ND;

  const int wgrid = (NC * ND + 255) / 256;
  wT_kernel<<<wgrid, 256, 0, stream>>>(Wq, wqt);
  wT_kernel<<<wgrid, 256, 0, stream>>>(Wk, wkt);
  wT_kernel<<<wgrid, 256, 0, stream>>>(Wv, wvt);

  // 2048 16-row tiles, 4 waves (tiles) per 128-thread block
  qkv_kernel<<<(NB * NT) / 64, 128, 0, stream>>>(x, wqt, bq, bk, bv, qb, kb, vtb);
  attn_kernel<<<(NB * NT) / 64, 128, 0, stream>>>(qb, kb, vtb, y);
}
